// MultiHeadAttention_55190329753910
// MI455X (gfx1250) — matrix operand order repaired and checked by reading, not compiled, not run on hardware
//
#include <hip/hip_runtime.h>
#include <math.h>

// ---------------- problem constants ----------------
#define D_MODEL 1024
#define NHEADS  16
#define DK      64
#define BATCH   2
#define SEQ     2048
#define BS      (BATCH * SEQ)   // 4096 rows total

typedef unsigned short u16b;
typedef __bf16 bf16;
typedef bf16  v16bf __attribute__((ext_vector_type(16)));
typedef bf16  v8bf  __attribute__((ext_vector_type(8)));
typedef float v8f   __attribute__((ext_vector_type(8)));
typedef unsigned int v4u __attribute__((ext_vector_type(4)));
typedef int v4i __attribute__((ext_vector_type(4)));
typedef int v8i __attribute__((ext_vector_type(8)));

union BF16x16 { v16bf v; v8bf h[2]; };

// round-to-nearest-even f32 -> bf16 (bit pattern as u16)
__device__ __forceinline__ u16b f2bf(float f) {
    unsigned u = __builtin_bit_cast(unsigned, f);
    unsigned r = u + 0x7FFFu + ((u >> 16) & 1u);
    return (u16b)(r >> 16);
}

// A-fragment (16x32 bf16, M x K): lane ln holds row M=ln.
// lo-half lanes: K {k0..k0+7, k0+16..k0+23}; hi-half: K {k0+8..k0+15, k0+24..k0+31}
__device__ __forceinline__ v16bf load_a_frag(const u16b* base, int row, int ld,
                                             int k0, int half) {
    BF16x16 u;
    const u16b* p = base + (size_t)row * ld + k0 + half * 8;
    u.h[0] = *reinterpret_cast<const v8bf*>(p);
    u.h[1] = *reinterpret_cast<const v8bf*>(p + 16);
    return u.v;
}

// B-fragment (32x16 bf16, K x N): lane ln holds column N=ln.
// Bt is stored transposed (N-major, K contiguous), so column = contiguous row.
__device__ __forceinline__ v16bf load_b_frag(const u16b* baseT, int col, int ld,
                                             int k0, int half) {
    BF16x16 u;
    const u16b* p = baseT + (size_t)col * ld + k0 + half * 8;
    u.h[0] = *reinterpret_cast<const v8bf*>(p);
    u.h[1] = *reinterpret_cast<const v8bf*>(p + 16);
    return u.v;
}

// ---------------- conversion kernels ----------------
__global__ void mha_cvt_bf16(const float* __restrict__ x, u16b* __restrict__ y, int n) {
    int i = blockIdx.x * 256 + threadIdx.x;
    if (i < n) y[i] = f2bf(x[i]);
}

// Wt[n][k] = bf16(W[k][n])  (D_MODEL x D_MODEL)
__global__ void mha_cvt_transpose_w(const float* __restrict__ W, u16b* __restrict__ Wt) {
    int idx = blockIdx.x * 256 + threadIdx.x;          // coalesced read along n
    int k = idx >> 10, n = idx & (D_MODEL - 1);
    Wt[(size_t)n * D_MODEL + k] = f2bf(W[idx]);
}

// ---------------- register-tiled bf16 WMMA GEMM ----------------
// C[M x N] = A[M x K](bf16) * Bt[N x K]^T(bf16) + bias
// Wave computes a 32x64 tile (2x4 grid of 16x16 WMMA tiles);
// block = 8 waves arranged 2(M) x 4(N) -> 64 x 256 per block.
// mode 0: bf16 row-major out      (Q, K projections)
// mode 1: bf16 head-transposed V  (Vt[(b*H+h)*DK + d][SEQ])
// mode 2: f32 row-major out       (final output projection)
__global__ __launch_bounds__(256)
void mha_gemm_bf16(const u16b* __restrict__ A, const u16b* __restrict__ Bt,
                   const float* __restrict__ bias,
                   u16b* __restrict__ outB, float* __restrict__ outF,
                   int M, int N, int K, int mode) {
    const int wave = threadIdx.x >> 5;
    const int lane = threadIdx.x & 31;
    const int half = lane >> 4;
    const int ln   = lane & 15;
    const int wm   = wave >> 2, wn = wave & 3;

    const int m_base = blockIdx.y * 64 + wm * 32;
    const int n_base = blockIdx.x * 256 + wn * 64;

    v8f acc[2][4] = {};
    for (int k0 = 0; k0 < K; k0 += 32) {
        // speculative prefetch of next k-step (safe past end; dropped on fault)
        __builtin_prefetch(A + (size_t)(m_base + ln) * K + k0 + 32, 0, 1);
        __builtin_prefetch(Bt + (size_t)(n_base + ln) * K + k0 + 32, 0, 1);
        v16bf a[2], b[4];
#pragma unroll
        for (int i = 0; i < 2; ++i)
            a[i] = load_a_frag(A, m_base + i * 16 + ln, K, k0, half);
#pragma unroll
        for (int j = 0; j < 4; ++j)
            b[j] = load_b_frag(Bt, n_base + j * 16 + ln, K, k0, half);
#pragma unroll
        for (int i = 0; i < 2; ++i)
#pragma unroll
            for (int j = 0; j < 4; ++j)
                acc[i][j] = __builtin_amdgcn_wmma_f32_16x16x32_bf16(
                    false, a[i], false, b[j], (short)0, acc[i][j], false, false);
    }

#pragma unroll
    for (int j = 0; j < 4; ++j) {
        const int n = n_base + j * 16 + ln;
        const float bv = bias ? bias[n] : 0.0f;
#pragma unroll
        for (int i = 0; i < 2; ++i) {
#pragma unroll
            for (int r = 0; r < 8; ++r) {
                const int m = m_base + i * 16 + r + half * 8;
                const float v = acc[i][j][r] + bv;
                if (mode == 0) {
                    outB[(size_t)m * N + n] = f2bf(v);
                } else if (mode == 1) {
                    const int bb = m >> 11, s = m & (SEQ - 1);      // m = b*SEQ + s
                    const int h = n >> 6,  d = n & (DK - 1);
                    outB[(((size_t)((bb * NHEADS + h) * DK + d)) << 11) + s] = f2bf(v);
                } else {
                    outF[(size_t)m * N + n] = v;
                }
            }
        }
    }
}

// ---------------- fused attention: scores -> softmax -> P@V ----------------
// One workgroup owns a 16-query panel for one (b,h):
//   phase 0: TDM DMA of the 16x64 bf16 Q tile into LDS (tensor_load_to_lds)
//   phase 1: scores 16x2048 via WMMA -> LDS (f32, 128 KB)
//   phase 2: row softmax in registers; write probs (f32, HBM, once);
//            pack normalized P to bf16 in place inside each row's LDS region
//   phase 3: attn = P @ V via WMMA, A-fragments straight from LDS,
//            2-way K-split across waves, reduced through LDS
__global__ __launch_bounds__(256)
void mha_attn_fused(const u16b* __restrict__ Qb, const u16b* __restrict__ Kb,
                    const u16b* __restrict__ Vt, float* __restrict__ probs,
                    u16b* __restrict__ attn) {
    __shared__ float sc[16][SEQ];          // 128 KB: scores, then packed bf16 P
    __shared__ float red[2][16][DK];       // 8 KB: PV k-split partials
    __shared__ u16b  qstage[16 * DK];      // 2 KB: TDM-staged Q tile (row-major, ld=DK)

    const int wave = threadIdx.x >> 5;
    const int lane = threadIdx.x & 31;
    const int half = lane >> 4;
    const int ln   = lane & 15;
    const int qt = blockIdx.x;
    const int bh = blockIdx.y;
    const int b  = bh >> 4, h = bh & 15;

    const u16b* qbase = Qb + (size_t)b * SEQ * D_MODEL + h * DK;
    const u16b* kbase = Kb + (size_t)b * SEQ * D_MODEL + h * DK;

    // ---- phase 0: TDM load of Q tile (16 rows x 64 cols, 2B elems) ----
    if (wave == 0) {
        const unsigned long long gaddr =
            (unsigned long long)(const void*)(qbase + (size_t)qt * 16 * D_MODEL);
        const unsigned lds_off = (unsigned)(unsigned long long)(const void*)&qstage[0];
        v4u g0;
        g0.x = 1u;                                        // count=1 (valid user D#)
        g0.y = lds_off;                                   // lds_addr
        g0.z = (unsigned)(gaddr & 0xFFFFFFFFu);           // global_addr[31:0]
        g0.w = (unsigned)((gaddr >> 32) & 0x1FFFFFFu) | (2u << 30);  // [56:32] | type=2
        v8i g1;
        g1[0] = 0x00010000;                               // data_size=1 (2 bytes)
        g1[1] = (int)((D_MODEL & 0xFFFF) << 16);          // tensor_dim0 lo16
        g1[2] = (int)((D_MODEL >> 16) | ((SEQ & 0xFFFF) << 16));  // dim0 hi | dim1 lo
        g1[3] = (int)((SEQ >> 16) | (DK << 16));          // dim1 hi | tile_dim0=64
        g1[4] = 16;                                       // tile_dim1=16, tile_dim2=0
        g1[5] = D_MODEL;                                  // tensor_dim0_stride lo32
        g1[6] = 0;                                        // stride0 hi | stride1 lo
        g1[7] = 0;                                        // stride1 hi
        const v4i gz = {0, 0, 0, 0};
#if defined(__clang_major__) && (__clang_major__ >= 23)
        const v8i gz8 = {0, 0, 0, 0, 0, 0, 0, 0};
        __builtin_amdgcn_tensor_load_to_lds(g0, g1, gz, gz, gz8, 0);
#else
        __builtin_amdgcn_tensor_load_to_lds(g0, g1, gz, gz, 0);
#endif
        __builtin_amdgcn_s_wait_tensorcnt(0);
    }
    __syncthreads();

    // ---- phase 1: scores panel; each wave covers 256 key columns ----
    const v16bf aq0 = load_a_frag(qstage, ln, DK, 0, half);   // ds loads
    const v16bf aq1 = load_a_frag(qstage, ln, DK, 32, half);
    for (int t = 0; t < 16; ++t) {
        const int kkt = wave * 16 + t;
        const v16bf b0 = load_b_frag(kbase, kkt * 16 + ln, D_MODEL, 0, half);
        const v16bf b1 = load_b_frag(kbase, kkt * 16 + ln, D_MODEL, 32, half);
        v8f acc = {};
        acc = __builtin_amdgcn_wmma_f32_16x16x32_bf16(false, aq0, false, b0,
                                                      (short)0, acc, false, false);
        acc = __builtin_amdgcn_wmma_f32_16x16x32_bf16(false, aq1, false, b1,
                                                      (short)0, acc, false, false);
#pragma unroll
        for (int r = 0; r < 8; ++r)
            sc[r + half * 8][kkt * 16 + ln] = acc[r] * 0.125f;   // 1/sqrt(64)
    }
    __syncthreads();

    // ---- phase 2: softmax; each wave owns 2 rows, 64 values per lane ----
    float* gprobs = probs + (size_t)bh * SEQ * SEQ + (size_t)qt * 16 * SEQ;
    for (int rr = 0; rr < 2; ++rr) {
        const int row = wave * 2 + rr;
        const float* src = &sc[row][lane * 64];
        float v[64];
        float m = -INFINITY;
#pragma unroll
        for (int i = 0; i < 64; i += 4) {
            float4 f4 = *reinterpret_cast<const float4*>(src + i);
            v[i] = f4.x; v[i + 1] = f4.y; v[i + 2] = f4.z; v[i + 3] = f4.w;
            m = fmaxf(m, fmaxf(fmaxf(f4.x, f4.y), fmaxf(f4.z, f4.w)));
        }
#pragma unroll
        for (int off = 16; off > 0; off >>= 1) m = fmaxf(m, __shfl_xor(m, off, 32));
        float s = 0.0f;
#pragma unroll
        for (int i = 0; i < 64; ++i) { v[i] = __expf(v[i] - m); s += v[i]; }
#pragma unroll
        for (int off = 16; off > 0; off >>= 1) s += __shfl_xor(s, off, 32);
        const float inv = 1.0f / s;

        float* gout = gprobs + (size_t)row * SEQ + lane * 64;
        // packed bf16 P stays inside this row's own LDS region:
        // pb(row, c) lives at u32 word row*2048 + c/2  (== u16 index row*2*SEQ + c)
        unsigned* pw = reinterpret_cast<unsigned*>(&sc[row][0]) + lane * 32;
#pragma unroll
        for (int i = 0; i < 64; i += 2) {
            const float x0 = v[i] * inv, x1 = v[i + 1] * inv;
            gout[i] = x0; gout[i + 1] = x1;
            pw[i >> 1] = ((unsigned)f2bf(x1) << 16) | (unsigned)f2bf(x0);
        }
    }
    __syncthreads();

    // ---- phase 3: attn = P @ V; waves = 4 d-tiles x 2 k-halves ----
    const u16b* pbase = reinterpret_cast<const u16b*>(&sc[0][0]); // pb(q,c)=pbase[q*2*SEQ+c]
    const u16b* vbase = Vt + (size_t)bh * DK * SEQ;
    const int dtile = wave & 3, kh = wave >> 2;
    v8f acc = {};
    for (int k0 = kh * 1024; k0 < kh * 1024 + 1024; k0 += 32) {
        BF16x16 ua;
        const u16b* pr = pbase + ln * (2 * SEQ) + k0 + half * 8;
        ua.h[0] = *reinterpret_cast<const v8bf*>(pr);        // ds_load_b128
        ua.h[1] = *reinterpret_cast<const v8bf*>(pr + 16);
        const v16bf bm = load_b_frag(vbase, dtile * 16 + ln, SEQ, k0, half);
        acc = __builtin_amdgcn_wmma_f32_16x16x32_bf16(false, ua.v, false, bm,
                                                      (short)0, acc, false, false);
    }
#pragma unroll
    for (int r = 0; r < 8; ++r)
        red[kh][r + half * 8][dtile * 16 + ln] = acc[r];
    __syncthreads();

    // reduce the 2 k-halves, write attn (bf16)
    const int e0 = threadIdx.x * 4;            // 16*64 elems, 4 per thread
#pragma unroll
    for (int e = e0; e < e0 + 4; ++e) {
        const int m = e >> 6, n = e & 63;
        const float vsum = red[0][m][n] + red[1][m][n];
        attn[(size_t)(b * SEQ + qt * 16 + m) * D_MODEL + h * DK + n] = f2bf(vsum);
    }
}

// ---------------- launch ----------------
extern "C" void kernel_launch(void* const* d_in, const int* in_sizes, int n_in,
                              void* d_out, int out_size, void* d_ws, size_t ws_size,
                              hipStream_t stream) {
    const float* q_in = (const float*)d_in[0];
    const float* k_in = (const float*)d_in[1];
    const float* v_in = (const float*)d_in[2];
    const float* Wq = (const float*)d_in[3];
    const float* bq = (const float*)d_in[4];
    const float* Wk = (const float*)d_in[5];
    const float* bk = (const float*)d_in[6];
    const float* Wv = (const float*)d_in[7];
    const float* bv = (const float*)d_in[8];
    const float* Wo = (const float*)d_in[9];
    const float* bo = (const float*)d_in[10];

    float* out   = (float*)d_out;                        // [BS, D_MODEL]
    float* probs = (float*)d_out + (size_t)BS * D_MODEL; // [B,H,S,S]

    // workspace carve-up (u16b elements); total 64 MB
    u16b* wqT = (u16b*)d_ws;
    u16b* wkT = wqT + (size_t)D_MODEL * D_MODEL;
    u16b* wvT = wkT + (size_t)D_MODEL * D_MODEL;
    u16b* wvT2= wvT;
    u16b* woT = wvT + (size_t)D_MODEL * D_MODEL;
    u16b* xq  = woT + (size_t)D_MODEL * D_MODEL;
    u16b* xk  = xq  + (size_t)BS * D_MODEL;
    u16b* xv  = xk  + (size_t)BS * D_MODEL;
    u16b* qb  = xv  + (size_t)BS * D_MODEL;
    u16b* kb  = qb  + (size_t)BS * D_MODEL;
    u16b* vt  = kb  + (size_t)BS * D_MODEL;   // [(b*H+h)*DK + d][SEQ]
    u16b* at  = vt  + (size_t)BS * D_MODEL;   // attn bf16 [BS, D_MODEL]
    (void)wvT2;

    const int nAct = BS * D_MODEL;            // 4,194,304
    mha_cvt_bf16<<<nAct / 256, 256, 0, stream>>>(q_in, xq, nAct);
    mha_cvt_bf16<<<nAct / 256, 256, 0, stream>>>(k_in, xk, nAct);
    mha_cvt_bf16<<<nAct / 256, 256, 0, stream>>>(v_in, xv, nAct);

    const int nW = D_MODEL * D_MODEL;         // 1,048,576
    mha_cvt_transpose_w<<<nW / 256, 256, 0, stream>>>(Wq, wqT);
    mha_cvt_transpose_w<<<nW / 256, 256, 0, stream>>>(Wk, wkT);
    mha_cvt_transpose_w<<<nW / 256, 256, 0, stream>>>(Wv, wvT);
    mha_cvt_transpose_w<<<nW / 256, 256, 0, stream>>>(Wo, woT);

    // projections: block tile 64(M) x 256(N)
    dim3 gp(D_MODEL / 256, BS / 64);
    mha_gemm_bf16<<<gp, 256, 0, stream>>>(xq, wqT, bq, qb, nullptr,
                                          BS, D_MODEL, D_MODEL, 0);
    mha_gemm_bf16<<<gp, 256, 0, stream>>>(xk, wkT, bk, kb, nullptr,
                                          BS, D_MODEL, D_MODEL, 0);
    mha_gemm_bf16<<<gp, 256, 0, stream>>>(xv, wvT, bv, vt, nullptr,
                                          BS, D_MODEL, D_MODEL, 1);

    // fused attention: one block per (16-query panel, head)
    dim3 gf(SEQ / 16, BATCH * NHEADS);
    mha_attn_fused<<<gf, 256, 0, stream>>>(qb, kb, vt, probs, at);

    // output projection -> f32
    mha_gemm_bf16<<<gp, 256, 0, stream>>>(at, woT, bo, nullptr, out,
                                          BS, D_MODEL, D_MODEL, 2);
}
